// MultiHeadedAttention_85280870630021
// MI455X (gfx1250) — matrix operand order repaired and checked by reading, not compiled, not run on hardware
//
#include <hip/hip_runtime.h>

#define D_MODEL 1024
#define NHEAD   16
#define DK      64
#define SEQ     2048
#define BATCH   2

typedef __bf16 bf16x16 __attribute__((ext_vector_type(16)));
typedef float  f32x8   __attribute__((ext_vector_type(8)));

union FragB {
    bf16x16        v;
    __bf16         b[16];
    unsigned short h[16];
    unsigned int   w[8];
    uint4          q[2];
};

// A-operand 16-bit layout: VGPR p holds K pair; lanes>=16 get +8 offset
__device__ __forceinline__ int kpair(int p, int hi8) {
    return (p < 4 ? 2 * p : 16 + 2 * (p - 4)) + hi8;
}

// CDNA5 async global->LDS copy (16 bytes per lane), tracked by ASYNCcnt.
// ldsoff: byte offset within the wave's LDS allocation (generic-ptr low 32 bits).
__device__ __forceinline__ void async_b128(unsigned ldsoff, const void* g) {
    asm volatile("global_load_async_to_lds_b128 %0, %1, off"
                 :: "v"(ldsoff), "v"(g) : "memory");
}
__device__ __forceinline__ void wait_async0() {
    asm volatile("s_wait_asynccnt 0" ::: "memory");
}

// ---------------------------------------------------------------------------
// One-shot: W (f32, row-major [o][d]) -> WT (bf16, [d][o])   (1M elements)
// ---------------------------------------------------------------------------
__global__ __launch_bounds__(256) void cvt_wT_kernel(
    const float* __restrict__ W, __bf16* __restrict__ WT)
{
    int i = blockIdx.x * 256 + threadIdx.x;     // WT flat index (coalesced store)
    int d = i >> 10, o = i & (D_MODEL - 1);
    WT[i] = (__bf16)W[(size_t)o * D_MODEL + d];
}

// ---------------------------------------------------------------------------
// Projection GEMM: y = (x_f32 @ WT + b) * scale, bf16 output.
// WT is pre-transposed bf16 [d][o]; staged by async b128, double-buffered.
// mode 0: store [B,H,S,64] (Q, V).  mode 1: store transposed [B,H,64,S] (K).
// ---------------------------------------------------------------------------
__global__ __launch_bounds__(128) void qkv_proj_kernel(
    const float* __restrict__ x, const __bf16* __restrict__ WT,
    const float* __restrict__ bias, __bf16* __restrict__ dst,
    float scale, int mode)
{
    __shared__ __align__(16) __bf16 sWT[2][32 * 64];   // [buf][d][o]
    const int tid  = threadIdx.x;
    const int lane = tid & 31, wave = tid >> 5;
    const int tbase = blockIdx.x * 64 + wave * 16;
    const int obase = blockIdx.y * 64;
    const int rowq = lane & 15;
    const int hi8  = (lane >> 4) << 3;   // 0 or 8
    const unsigned sWT_off = (unsigned)(size_t)&sWT[0][0];

    // stage one 32d x 64o tile of WT: 32 rows x 8 b128 segments = 256 transfers
    auto stage = [&](int kc, int buf) {
#pragma unroll
        for (int r = 0; r < 2; ++r) {
            int t = tid + r * 128;               // 0..255
            int d = t >> 3, seg = t & 7;
            async_b128(sWT_off + (unsigned)buf * (32 * 64 * 2) + t * 16,
                       WT + (size_t)(kc * 32 + d) * D_MODEL + obase + seg * 8);
        }
    };

    f32x8 acc[4];
#pragma unroll
    for (int s = 0; s < 4; ++s)
#pragma unroll
        for (int j = 0; j < 8; ++j) acc[s][j] = 0.0f;

    stage(0, 0);
    wait_async0();
    __syncthreads();

    for (int kc = 0; kc < D_MODEL / 32; ++kc) {
        const int cur = kc & 1;
        if (kc + 1 < D_MODEL / 32) stage(kc + 1, cur ^ 1);   // overlap copy/compute

        // A fragment: 16 token rows x 32 K, f32 -> bf16 on the fly
        FragB a;
        const float* xr = x + (size_t)(tbase + rowq) * D_MODEL + kc * 32;
#pragma unroll
        for (int p = 0; p < 8; ++p) {
            int k = kpair(p, hi8);
            a.b[2 * p]     = (__bf16)xr[k];
            a.b[2 * p + 1] = (__bf16)xr[k + 1];
        }
#pragma unroll
        for (int s = 0; s < 4; ++s) {
            FragB bf;
#pragma unroll
            for (int e = 0; e < 16; ++e)
                bf.b[e] = sWT[cur][((e < 8 ? e : e + 8) + hi8) * 64 + s * 16 + rowq];
            acc[s] = __builtin_amdgcn_wmma_f32_16x16x32_bf16(
                false, a.v, false, bf.v, (short)0, acc[s], false, false);
        }
        wait_async0();       // next buffer landed
        __syncthreads();     // and everyone finished reading current buffer
    }

    // epilogue: bias, scale, bf16 store
#pragma unroll
    for (int s = 0; s < 4; ++s) {
        int col = obase + s * 16 + rowq;
        float bv = bias[col];
        int h = col >> 6, d = col & 63;
#pragma unroll
        for (int j = 0; j < 8; ++j) {
            int t  = tbase + hi8 + j;
            int bb = t >> 11, sq = t & (SEQ - 1);
            size_t off = (mode == 0)
                ? ((size_t)(bb * NHEAD + h) * SEQ + sq) * DK + d
                : ((size_t)(bb * NHEAD + h) * DK + d) * SEQ + sq;
            dst[off] = (__bf16)((acc[s][j] + bv) * scale);
        }
    }
}

// ---------------------------------------------------------------------------
// Flash attention. Q,V in [B,H,S,64]; K pre-transposed [B,H,64,S].
// Per block: one (b,h), 64 query rows; K loop in chunks of 32 columns.
// K^T and V chunks async-staged to LDS, double-buffered against compute.
// ---------------------------------------------------------------------------
__global__ __launch_bounds__(128) void flash_attn_kernel(
    const __bf16* __restrict__ Qb, const __bf16* __restrict__ Ktb,
    const __bf16* __restrict__ Vb, __bf16* __restrict__ Xb,
    const unsigned char* __restrict__ amask, const unsigned char* __restrict__ kpmask)
{
    __shared__ __align__(16) __bf16 sKT[2][64 * 32];     // [buf][d][k]
    __shared__ __align__(16) __bf16 sV [2][32 * 64];     // [buf][k][d]
    __shared__ __align__(16) __bf16 sP [4][16 * 32];     // per-wave P [q][k]

    const int tid  = threadIdx.x;
    const int lane = tid & 31, wave = tid >> 5;
    const int bh   = blockIdx.y;            // b*NHEAD + h
    const int b    = bh >> 4;
    const int h    = bh & (NHEAD - 1);
    const int qbase = blockIdx.x * 64 + wave * 16;
    const int rowq = lane & 15;
    const int hi8  = (lane >> 4) << 3;

    const size_t headoff = (size_t)bh * SEQ * DK;   // same size for Kt layout
    const unsigned sKT_off = (unsigned)(size_t)&sKT[0][0];
    const unsigned sV_off  = (unsigned)(size_t)&sV[0][0];

    auto stage = [&](int it, int buf) {
        const int kbase = it * 32;
        const unsigned ko = sKT_off + (unsigned)buf * (64 * 32 * 2);
        const unsigned vo = sV_off  + (unsigned)buf * (32 * 64 * 2);
#pragma unroll
        for (int r = 0; r < 2; ++r) {
            int t = tid + r * 128;               // 0..255
            int d = t >> 2, seg = t & 3;         // Kt slice: 64 rows x 4 b128
            async_b128(ko + t * 16, Ktb + headoff + (size_t)d * SEQ + kbase + seg * 8);
            async_b128(vo + t * 16, Vb  + headoff + (size_t)kbase * DK + t * 8);
        }
    };

    // Q fragments (already scaled by 1/sqrt(dk)); 2 chunks of K=32
    FragB qa[2];
    {
        const __bf16* qr = Qb + headoff + (size_t)(qbase + rowq) * DK;
#pragma unroll
        for (int c = 0; c < 2; ++c)
#pragma unroll
            for (int p = 0; p < 8; ++p) {
                int k = c * 32 + kpair(p, hi8);
                qa[c].w[p] = *(const unsigned int*)(qr + k);
            }
    }

    f32x8 acc[4];
    float m[8], l[8];
#pragma unroll
    for (int s = 0; s < 4; ++s)
#pragma unroll
        for (int j = 0; j < 8; ++j) acc[s][j] = 0.0f;
#pragma unroll
    for (int j = 0; j < 8; ++j) { m[j] = -1e30f; l[j] = 0.0f; }

    stage(0, 0);
    wait_async0();
    __syncthreads();

    for (int it = 0; it < SEQ / 32; ++it) {
        const int kbase = it * 32;
        const int cur = it & 1;
        if (it + 1 < SEQ / 32) {
            stage(it + 1, cur ^ 1);              // overlap copy with compute
            __builtin_prefetch(Vb + headoff + (size_t)(kbase + 64) * DK + tid * 16, 0, 1);
        }
        const __bf16* KT = sKT[cur];
        const __bf16* VS = sV[cur];

        // ---- scores: two 16x16 tiles, each accumulated over d = 2x32 ----
        f32x8 sc[2];
#pragma unroll
        for (int ks2 = 0; ks2 < 2; ++ks2) {
#pragma unroll
            for (int j = 0; j < 8; ++j) sc[ks2][j] = 0.0f;
#pragma unroll
            for (int c = 0; c < 2; ++c) {
                FragB kb;
#pragma unroll
                for (int e = 0; e < 16; ++e)
                    kb.b[e] = KT[(c * 32 + (e < 8 ? e : e + 8) + hi8) * 32 + ks2 * 16 + rowq];
                sc[ks2] = __builtin_amdgcn_wmma_f32_16x16x32_bf16(
                    false, qa[c].v, false, kb.v, (short)0, sc[ks2], false, false);
            }
            // masks (col fixed per lane; row per fragment register)
            int kcol = kbase + ks2 * 16 + rowq;
            bool kpm = kpmask[(size_t)b * SEQ + kcol] != 0;
#pragma unroll
            for (int j = 0; j < 8; ++j) {
                int qrow = qbase + hi8 + j;
                bool am = amask[((size_t)b * SEQ + qrow) * SEQ + kcol] != 0;
                sc[ks2][j] = (am || kpm) ? -1e30f : sc[ks2][j];
            }
        }

        // ---- online softmax (row stats replicated across 16-lane groups) ----
        __bf16* Pw = sP[wave];
#pragma unroll
        for (int j = 0; j < 8; ++j) {
            float t = fmaxf(sc[0][j], sc[1][j]);
            t = fmaxf(t, __shfl_xor(t, 1));
            t = fmaxf(t, __shfl_xor(t, 2));
            t = fmaxf(t, __shfl_xor(t, 4));
            t = fmaxf(t, __shfl_xor(t, 8));
            float mnew = fmaxf(m[j], t);
            float corr = __expf(m[j] - mnew);
            float p0 = (sc[0][j] > -1e29f) ? __expf(sc[0][j] - mnew) : 0.0f;
            float p1 = (sc[1][j] > -1e29f) ? __expf(sc[1][j] - mnew) : 0.0f;
            float rs = p0 + p1;
            rs += __shfl_xor(rs, 1);
            rs += __shfl_xor(rs, 2);
            rs += __shfl_xor(rs, 4);
            rs += __shfl_xor(rs, 8);
            l[j] = l[j] * corr + rs;
            m[j] = mnew;
#pragma unroll
            for (int s = 0; s < 4; ++s) acc[s][j] *= corr;
            // stash P (C-layout -> natural [q][k] in per-wave LDS)
            Pw[(hi8 + j) * 32 + rowq]      = (__bf16)p0;
            Pw[(hi8 + j) * 32 + 16 + rowq] = (__bf16)p1;
        }
        // same-wave DS ops are in-order; wait for write data, fence the compiler
        asm volatile("s_wait_dscnt 0" ::: "memory");

        // P A-fragment (16q x 32k)
        FragB pa;
#pragma unroll
        for (int p = 0; p < 8; ++p) {
            int k = kpair(p, hi8);
            pa.w[p] = *(const unsigned int*)&Pw[rowq * 32 + k];
        }
        // ---- P @ V: 4 WMMAs across d = 64 ----
#pragma unroll
        for (int s = 0; s < 4; ++s) {
            FragB vb;
#pragma unroll
            for (int e = 0; e < 16; ++e)
                vb.b[e] = VS[((e < 8 ? e : e + 8) + hi8) * 64 + s * 16 + rowq];
            acc[s] = __builtin_amdgcn_wmma_f32_16x16x32_bf16(
                false, pa.v, false, vb.v, (short)0, acc[s], false, false);
        }

        wait_async0();       // next buffer landed
        __syncthreads();     // everyone done reading current buffer
    }

    // normalize + store bf16 into concat-head layout [B*S][D_MODEL]
#pragma unroll
    for (int j = 0; j < 8; ++j) {
        float inv = (l[j] > 0.0f) ? (1.0f / l[j]) : 0.0f;
        int trow = b * SEQ + qbase + hi8 + j;
#pragma unroll
        for (int s = 0; s < 4; ++s) {
            int col = h * 64 + s * 16 + rowq;
            Xb[(size_t)trow * D_MODEL + col] = (__bf16)(acc[s][j] * inv);
        }
    }
}

// ---------------------------------------------------------------------------
// Output projection: out = Xb(bf16) @ WoT(bf16) + bo  (fp32 out)
// ---------------------------------------------------------------------------
__global__ __launch_bounds__(128) void out_proj_kernel(
    const __bf16* __restrict__ xb, const __bf16* __restrict__ WT,
    const float* __restrict__ bias, float* __restrict__ out)
{
    __shared__ __align__(16) __bf16 sWT[2][32 * 64];
    const int tid  = threadIdx.x;
    const int lane = tid & 31, wave = tid >> 5;
    const int tbase = blockIdx.x * 64 + wave * 16;
    const int obase = blockIdx.y * 64;
    const int rowq = lane & 15;
    const int hi8  = (lane >> 4) << 3;
    const unsigned sWT_off = (unsigned)(size_t)&sWT[0][0];

    auto stage = [&](int kc, int buf) {
#pragma unroll
        for (int r = 0; r < 2; ++r) {
            int t = tid + r * 128;
            int d = t >> 3, seg = t & 7;
            async_b128(sWT_off + (unsigned)buf * (32 * 64 * 2) + t * 16,
                       WT + (size_t)(kc * 32 + d) * D_MODEL + obase + seg * 8);
        }
    };

    f32x8 acc[4];
#pragma unroll
    for (int s = 0; s < 4; ++s)
#pragma unroll
        for (int j = 0; j < 8; ++j) acc[s][j] = 0.0f;

    stage(0, 0);
    wait_async0();
    __syncthreads();

    for (int kc = 0; kc < D_MODEL / 32; ++kc) {
        const int cur = kc & 1;
        if (kc + 1 < D_MODEL / 32) stage(kc + 1, cur ^ 1);

        FragB a;
        const __bf16* xr = xb + (size_t)(tbase + rowq) * D_MODEL + kc * 32;
#pragma unroll
        for (int p = 0; p < 8; ++p) {
            int k = kpair(p, hi8);
            a.w[p] = *(const unsigned int*)(xr + k);
        }
#pragma unroll
        for (int s = 0; s < 4; ++s) {
            FragB bf;
#pragma unroll
            for (int e = 0; e < 16; ++e)
                bf.b[e] = sWT[cur][((e < 8 ? e : e + 8) + hi8) * 64 + s * 16 + rowq];
            acc[s] = __builtin_amdgcn_wmma_f32_16x16x32_bf16(
                false, a.v, false, bf.v, (short)0, acc[s], false, false);
        }
        wait_async0();
        __syncthreads();
    }

#pragma unroll
    for (int s = 0; s < 4; ++s) {
        int col = obase + s * 16 + rowq;
        float bv = bias[col];
#pragma unroll
        for (int j = 0; j < 8; ++j) {
            int t = tbase + hi8 + j;
            out[(size_t)t * D_MODEL + col] = acc[s][j] + bv;
        }
    }
}

// ---------------------------------------------------------------------------
extern "C" void kernel_launch(void* const* d_in, const int* in_sizes, int n_in,
                              void* d_out, int out_size, void* d_ws, size_t ws_size,
                              hipStream_t stream) {
    (void)in_sizes; (void)n_in; (void)out_size; (void)ws_size;

    const float* query = (const float*)d_in[0];
    const float* key   = (const float*)d_in[1];
    const float* value = (const float*)d_in[2];
    const unsigned char* amask  = (const unsigned char*)d_in[3];
    const unsigned char* kpmask = (const unsigned char*)d_in[4];
    const float* Wq = (const float*)d_in[5];  const float* bq = (const float*)d_in[6];
    const float* Wk = (const float*)d_in[7];  const float* bk = (const float*)d_in[8];
    const float* Wv = (const float*)d_in[9];  const float* bv = (const float*)d_in[10];
    const float* Wo = (const float*)d_in[11]; const float* bo = (const float*)d_in[12];
    float* out = (float*)d_out;

    const size_t NTOK = (size_t)BATCH * SEQ;           // 4096
    const size_t WSZ  = (size_t)D_MODEL * D_MODEL;     // 1M
    __bf16* Qb  = (__bf16*)d_ws;                       // activations: 8 MiB each
    __bf16* Ktb = Qb  + NTOK * D_MODEL;                // K stored [B,H,64,S]
    __bf16* Vb  = Ktb + NTOK * D_MODEL;
    __bf16* Xb  = Vb  + NTOK * D_MODEL;
    __bf16* WTq = Xb  + NTOK * D_MODEL;                // weights: 2 MiB each
    __bf16* WTk = WTq + WSZ;
    __bf16* WTv = WTk + WSZ;
    __bf16* WTo = WTv + WSZ;

    dim3 blk(128);
    dim3 gcvt(WSZ / 256);                              // 4096 blocks
    dim3 gproj(NTOK / 64, D_MODEL / 64);               // 64 x 16
    dim3 gattn(SEQ / 64, BATCH * NHEAD);               // 32 x 32

    cvt_wT_kernel<<<gcvt, 256, 0, stream>>>(Wq, WTq);
    cvt_wT_kernel<<<gcvt, 256, 0, stream>>>(Wk, WTk);
    cvt_wT_kernel<<<gcvt, 256, 0, stream>>>(Wv, WTv);
    cvt_wT_kernel<<<gcvt, 256, 0, stream>>>(Wo, WTo);

    qkv_proj_kernel<<<gproj, blk, 0, stream>>>(query, WTq, bq, Qb,  0.125f, 0);
    qkv_proj_kernel<<<gproj, blk, 0, stream>>>(key,   WTk, bk, Ktb, 1.0f,   1);
    qkv_proj_kernel<<<gproj, blk, 0, stream>>>(value, WTv, bv, Vb,  1.0f,   0);
    flash_attn_kernel<<<gattn, blk, 0, stream>>>(Qb, Ktb, Vb, Xb, amask, kpmask);
    out_proj_kernel<<<gproj, blk, 0, stream>>>(Xb, WTo, bo, out);
}
